// ResNet_91311004713003
// MI455X (gfx1250) — compile-verified, matrix operand order checked by reading, not run on hardware
//
#include <hip/hip_runtime.h>

typedef __attribute__((ext_vector_type(16))) __bf16 v16bf;
typedef __attribute__((ext_vector_type(8)))  __bf16 v8bf;
typedef __attribute__((ext_vector_type(8)))  float  v8f;
typedef __attribute__((ext_vector_type(4)))  int    v4i;

// ---- problem constants ----
#define NBATCH 4
#define CIN    5
#define DD     16
#define HH     224
#define WIN    224
#define CO     64
#define OD     16
#define OH     112
#define OW     112
#define SPATIAL   802816     // NBATCH*16*112*112
#define PER_BATCH 200704     // 16*112*112
#define PER_OD    12544      // 112*112
// padded input (all gathers in bounds): id'=od+kd (0..21), ih'=2oh+kh (0..228),
// iw'=2ow+kw (0..229, 8-wide rows)
#define PD     22
#define PH     230
#define PW     232           // multiple of 8
#define NPAD   (NBATCH*CIN*PD*PH*PW)   // 23,478,400 halfs (~47 MB, L2-resident)
// K layout: 245 (ci,kd,kh) rows x 8 kw-slots (slot 7 = zero weight),
// padded to 248 rows -> 62 chunks of 32.
#define KROWS  245
#define KPAD   1984
#define NCHUNK 62
#define BPITCH 40            // LDS pitch in halfs (80 B, 16B-aligned vectors)
#define BUFH   (64 * BPITCH) // one ping-pong buffer, in halfs

// ---- CDNA5 async global->LDS path (ASYNCcnt), with compile-safe fallback ----
#if __has_builtin(__builtin_amdgcn_global_load_async_to_lds_b128)
#define HAVE_ASYNC 1
#else
#define HAVE_ASYNC 0
#endif

__device__ __forceinline__ void stage16(const unsigned short* src,
                                        unsigned short* dst) {
#if HAVE_ASYNC
  __builtin_amdgcn_global_load_async_to_lds_b128(
      (__attribute__((address_space(1))) v4i*)src,
      (__attribute__((address_space(3))) v4i*)dst, 0, 0);
#else
  *(uint4*)dst = *(const uint4*)src;   // global_load_b128 + ds_store_b128
#endif
}

__device__ __forceinline__ void stage_wait() {
#if HAVE_ASYNC
#if __has_builtin(__builtin_amdgcn_s_wait_asynccnt)
  __builtin_amdgcn_s_wait_asynccnt(0);
#else
  asm volatile("s_wait_asynccnt 0x0" ::: "memory");
#endif
#endif
}

// Closed-form equivalent of the 128-pass posit bucket loop.
__device__ __forceinline__ float posit_q(float x) {
  unsigned u = __float_as_uint(x);
  unsigned a = u & 0x7fffffffu;
  unsigned sgn = u & 0x80000000u;
  if (a == 0u) return x;
  if (a < 0x37880000u)                       // 0 < |x| < 1.0625*2^-16 -> 2^-16
    return __uint_as_float(sgn | 0x37800000u);
  int p = (int)(a >> 23) - 127;              // floor(log2|x|)
  float f = __uint_as_float((a & 0x007fffffu) | 0x3f800000u);  // [1,2)
  float q;
  if (p >= 0) {                              // |x| >= 1 regime (g1 table)
    float jf = rintf(8.0f * (f - 1.0f));
    if (f > 1.875f) { jf = 0.0f; p += 1; }   // later octave wins the overlap
    if (p > 15) return x;                    // beyond table: keep original
    q = (1.0f + 0.125f * jf) * __uint_as_float((unsigned)(127 + p) << 23);
  } else {                                   // |x| < 1 regime (l1 table)
    if (f < 1.0625f || f > 1.9375f) return x;  // dead zones keep original
    float jf = rintf(8.0f * (f - 1.0f));       // 1..7
    q = (1.0f + 0.125f * jf) * __uint_as_float((unsigned)(127 + p) << 23);
  }
  return __uint_as_float(sgn | (__float_as_uint(q) & 0x7fffffffu));
}

__device__ __forceinline__ unsigned short f2bf(float v) {   // RNE f32->bf16
  unsigned b = __float_as_uint(v);
  b += 0x7fffu + ((b >> 16) & 1u);
  return (unsigned short)(b >> 16);
}

// W [64][1715] fp32 -> Wq [64][1984] bf16; k = row*8 + kw, row=(ci,kd,kh),
// kw==7 slots and rows>=245 are zero (they absorb the 8-wide gather junk).
__global__ void quantW_kernel(const float* __restrict__ W,
                              unsigned short* __restrict__ Wq) {
  int i = blockIdx.x * blockDim.x + threadIdx.x;
  if (i >= CO * KPAD) return;
  int co = i / KPAD, k = i - co * KPAD;
  int row = k >> 3, kw = k & 7;
  unsigned short v = 0;
  if (row < KROWS && kw < 7) {
    int ci = row / 49; int rr = row - ci * 49;
    int kd = rr / 7;   int kh = rr - kd * 7;
    v = f2bf(posit_q(W[co * 1715 + ci * 343 + kd * 49 + kh * 7 + kw]));
  }
  Wq[i] = v;
}

// x fp32 -> posit-quantized bf16 into zero-padded layout [4][5][22][230][232].
__global__ void quantXpad_kernel(const float* __restrict__ x,
                                 unsigned short* __restrict__ xqp) {
  int i = blockIdx.x * blockDim.x + threadIdx.x;
  if (i >= NPAD) return;
  int iw = i % PW; int t = i / PW;
  int ih = t % PH; t /= PH;
  int id = t % PD; t /= PD;
  int ci = t % CIN; int nb = t / CIN;
  int sx = iw - 3, sy = ih - 3, sz = id - 3;
  unsigned short v = 0;
  if ((unsigned)sz < (unsigned)DD && (unsigned)sy < (unsigned)HH &&
      (unsigned)sx < (unsigned)WIN)
    v = f2bf(posit_q(x[(((nb * CIN + ci) * DD + sz) * HH + sy) * WIN + sx]));
  xqp[i] = v;
}

// Implicit-GEMM conv: block = 64 couts x 64 positions, 8 waves,
// wave = 16 couts x 32 positions = 2 x v_wmma_f32_16x16x32_bf16 per K-chunk.
// B-staging: async global->LDS b128 per (position, kw-row), ping-pong buffered
// so chunk c+1's gather overlaps chunk c's WMMAs; one barrier per chunk.
__global__ void __launch_bounds__(256)
conv_kernel(const unsigned short* __restrict__ Wq,
            const unsigned short* __restrict__ xqp,
            float* __restrict__ out) {
  __shared__ __align__(16) unsigned short Bs[2 * BUFH];  // 10 KB

  const int tid  = threadIdx.x;
  const int lane = tid & 31;
  const int wave = tid >> 5;
  const int mtile = wave & 3;        // cout quarter
  const int nhalf = wave >> 2;       // position half (32 each)
  const int pbase = blockIdx.x * 64; // first linear output position

  // ---- staging role: fixed row-in-chunk, fixed position ----
  const int rsub = tid & 3;          // kw-row within chunk (0..3)
  const int spos = tid >> 2;         // position 0..63
  int lin = pbase + spos;
  int nb = lin / PER_BATCH; int r0 = lin - nb * PER_BATCH;
  int od = r0 / PER_OD;     r0 -= od * PER_OD;
  int oh = r0 / OW;         int ow = r0 - oh * OW;
  const int oh2 = 2 * oh, ow2 = 2 * ow;
  const int nbci = nb * CIN;
  unsigned short* const dst0 = Bs + spos * BPITCH + rsub * 8;

  auto issue = [&](int cc) {
    int row = cc * 4 + rsub;
    if (row >= KROWS) row = 0;               // padded rows: weights are zero
    int ci = row / 49; int rr = row - ci * 49;
    int kd = rr / 7;   int kh = rr - kd * 7;
    const unsigned short* src = xqp +
        ((((nbci + ci) * PD + (od + kd)) * PH + (oh2 + kh)) * PW + ow2);
    stage16(src, dst0 + (cc & 1) * BUFH);
  };

  v8f acc[2] = {};
  const int l15 = lane & 15, lh = lane >> 4;
  const unsigned short* arow = Wq + (mtile * 16 + l15) * KPAD;

  issue(0);
  for (int c = 0; c < NCHUNK; ++c) {
    stage_wait();                 // my chunk-c gather has landed in LDS
    __syncthreads();              // everyone's has; all reads of buf[c-1] done
    if (c + 1 < NCHUNK) issue(c + 1);   // overlaps with this chunk's WMMAs

    const int kb = c * 32;
    __builtin_prefetch(arow + kb + 32, 0, 3);

    // ---- A fragment straight from L2-resident Wq ----
    const v8bf alo = *(const v8bf*)(arow + kb + lh * 8);
    const v8bf ahi = *(const v8bf*)(arow + kb + 16 + lh * 8);
    const v16bf Af = __builtin_shufflevector(alo, ahi,
        0, 1, 2, 3, 4, 5, 6, 7, 8, 9, 10, 11, 12, 13, 14, 15);

    // ---- two 16x16x32 WMMAs over this wave's 32 positions ----
    const unsigned short* bbuf = Bs + (c & 1) * BUFH;
#pragma unroll
    for (int s = 0; s < 2; ++s) {
      const unsigned short* bp =
          bbuf + (nhalf * 32 + s * 16 + l15) * BPITCH + lh * 8;
      const v8bf blo = *(const v8bf*)(bp);
      const v8bf bhi = *(const v8bf*)(bp + 16);
      const v16bf Bf = __builtin_shufflevector(blo, bhi,
          0, 1, 2, 3, 4, 5, 6, 7, 8, 9, 10, 11, 12, 13, 14, 15);
      acc[s] = __builtin_amdgcn_wmma_f32_16x16x32_bf16(
          false, Af, false, Bf, (short)0, acc[s], false, false);
    }
  }

  // ---- write D: lane l holds N=(l&15), M=(l>>4)*8 + r ----
#pragma unroll
  for (int s = 0; s < 2; ++s) {
    int pos = pbase + nhalf * 32 + s * 16 + l15;
    int pnb = pos / PER_BATCH; int pr = pos - pnb * PER_BATCH;
    int pod = pr / PER_OD;     int sp = pr - pod * PER_OD;
#pragma unroll
    for (int rr2 = 0; rr2 < 8; ++rr2) {
      int co = mtile * 16 + lh * 8 + rr2;
      out[((pnb * CO + co) * DD + pod) * PER_OD + sp] = acc[s][rr2];
    }
  }
}

extern "C" void kernel_launch(void* const* d_in, const int* in_sizes, int n_in,
                              void* d_out, int out_size, void* d_ws, size_t ws_size,
                              hipStream_t stream) {
  (void)n_in; (void)out_size; (void)ws_size; (void)in_sizes;
  const float* x = (const float*)d_in[0];
  const float* W = (const float*)d_in[1];
  float* out = (float*)d_out;

  unsigned short* Wq  = (unsigned short*)d_ws;   // 64*1984 halfs (254 KB)
  unsigned short* xqp = Wq + CO * KPAD;          // 23,478,400 halfs (~47 MB)

  quantW_kernel<<<(CO * KPAD + 255) / 256, 256, 0, stream>>>(W, Wq);
  quantXpad_kernel<<<(NPAD + 255) / 256, 256, 0, stream>>>(x, xqp);
  conv_kernel<<<SPATIAL / 64, 256, 0, stream>>>(Wq, xqp, out);
}
